// MetaWeightNet_61014305407329
// MI455X (gfx1250) — matrix operand order repaired and checked by reading, not compiled, not run on hardware
//
#include <hip/hip_runtime.h>

#define S_TOTAL 262144
#define HDIM    64
#define NUSERS  1000000
#define ROWS    128
#define NBLK    (S_TOTAL / ROWS)   // 2048
#define A_STRIDE 200               // 192 bf16 + 8 pad (400 B row stride, 16B-aligned)
#define W_STRIDE 200
#define BN_EPS  1e-5f

typedef __attribute__((ext_vector_type(16))) __bf16 v16bf;
typedef __attribute__((ext_vector_type(8)))  __bf16 v8bf;
typedef __attribute__((ext_vector_type(8)))  float  v8f;

__device__ __forceinline__ unsigned short f2bf(float x) {
  __bf16 h = (__bf16)x;
  return __builtin_bit_cast(unsigned short, h);
}

// ---------------------------------------------------------------------------
// Kernel 1: convert the two 96x192 f32 weight matrices to bf16 in workspace.
// ---------------------------------------------------------------------------
__global__ __launch_bounds__(256) void wconv(const float* __restrict__ w_ssl,
                                             const float* __restrict__ w_rs,
                                             unsigned short* __restrict__ o_ssl,
                                             unsigned short* __restrict__ o_rs) {
  int i = blockIdx.x * 256 + threadIdx.x;   // 72 blocks * 256 = 18432 = 96*192
  if (i < 96 * 192) {
    o_ssl[i] = f2bf(w_ssl[i]);
    o_rs[i]  = f2bf(w_rs[i]);
  }
}

// ---------------------------------------------------------------------------
// Kernel 2: per (branch,beh) slice: gather -> bf16 LDS tile -> WMMA GEMM ->
// fused W2 dot + x3 dot + BN partial sums.  grid=(NBLK, 6), block=256 (8 waves)
// ---------------------------------------------------------------------------
__global__ __launch_bounds__(256) void mwn_main(
    const float* __restrict__ infoNCE, const float* __restrict__ blossm,
    const int*   __restrict__ stepidx, const int*   __restrict__ uidxl,
    const float* __restrict__ uembs,   const float* __restrict__ uemb,
    const unsigned short* __restrict__ Wbf_ssl,
    const unsigned short* __restrict__ Wbf_rs,
    const float* __restrict__ b_ssl1, const float* __restrict__ W_ssl2,
    const float* __restrict__ b_ssl2, const float* __restrict__ W_ssl3,
    const float* __restrict__ b_ssl3,
    const float* __restrict__ b_rs1,  const float* __restrict__ W_rs2,
    const float* __restrict__ b_rs2,  const float* __restrict__ W_rs3,
    const float* __restrict__ b_rs3,
    const float* __restrict__ prelu_ap,
    float* __restrict__ xvals, float4* __restrict__ partials) {

  __shared__ __align__(16) __bf16 sA[ROWS * A_STRIDE];  // 51200 B
  __shared__ __align__(16) __bf16 sW[96 * W_STRIDE];    // 38400 B
  __shared__ float x1buf[ROWS];
  __shared__ float x3buf[ROWS];
  __shared__ float red4[4][4];

  const int bx = blockIdx.x;
  const int by = blockIdx.y;            // 0..5
  const int branch = by / 3;            // 0 = SSL, 1 = RS
  const int beh    = by % 3;
  const int tid = threadIdx.x;
  const float a_pr = prelu_ap[0];

  const float* lossp; const unsigned short* Wbf;
  const float *b1, *W2, *W3; float b2, b3;
  if (branch == 0) {
    lossp = infoNCE + (size_t)beh * S_TOTAL;
    Wbf = Wbf_ssl; b1 = b_ssl1; W2 = W_ssl2; b2 = b_ssl2[0]; W3 = W_ssl3; b3 = b_ssl3[0];
  } else {
    lossp = blossm + (size_t)beh * S_TOTAL;
    Wbf = Wbf_rs;  b1 = b_rs1;  W2 = W_rs2;  b2 = b_rs2[0];  W3 = W_rs3;  b3 = b_rs3[0];
  }

  // ---- stage bf16 W1 (96 x 192, column(n)-major over k) into padded LDS ----
  {
    const unsigned int* wsrc = (const unsigned int*)Wbf;   // 2 bf16 per dword
    unsigned int* sWu = (unsigned int*)sW;
    for (int i = tid; i < 96 * 96; i += 256) {             // 96 dwords per row
      int r = i / 96, c = i % 96;
      sWu[r * (W_STRIDE / 2) + c] = wsrc[r * 96 + c];
    }
  }

  // ---- gather phase: 2 threads per row, build A tile + fused x3 dot ----
  {
    const int rl = tid >> 1;                 // local row 0..127
    const int h  = tid & 1;                  // half (32 floats each)
    const int grow = bx * ROWS + rl;
    const float loss = lossp[grow];

    const float *e1p, *e2p;
    if (branch == 0) {
      int u = stepidx[grow];
      e1p = uembs + ((size_t)beh * NUSERS + (size_t)u) * HDIM;  // ue_i[step]
      e2p = uemb  + (size_t)u * HDIM;                           // u[step]
    } else {
      int u = uidxl[(size_t)beh * S_TOTAL + grow];
      e1p = uemb  + (size_t)u * HDIM;                           // u_r
      e2p = uembs + ((size_t)beh * NUSERS + (size_t)u) * HDIM;  // ue_i_r
    }

    // loss broadcast into columns [0,64)
    {
      unsigned int lp = (unsigned int)f2bf(loss) * 0x00010001u;
      unsigned int* lrow = (unsigned int*)(sA + (size_t)rl * A_STRIDE + h * 32);
      #pragma unroll
      for (int j = 0; j < 16; ++j) lrow[j] = lp;
    }

    float dot = 0.f;
    // e1 -> columns [64,128), contributes W3[0..63]
    {
      const float4* e4 = (const float4*)(e1p + h * 32);
      const float4* w4 = (const float4*)(W3 + h * 32);
      unsigned int* drow = (unsigned int*)(sA + (size_t)rl * A_STRIDE + 64 + h * 32);
      #pragma unroll
      for (int q = 0; q < 8; ++q) {
        float4 v = e4[q]; float4 w = w4[q];
        dot += v.x * w.x + v.y * w.y + v.z * w.z + v.w * w.w;
        drow[q * 2 + 0] = (unsigned int)f2bf(v.x) | ((unsigned int)f2bf(v.y) << 16);
        drow[q * 2 + 1] = (unsigned int)f2bf(v.z) | ((unsigned int)f2bf(v.w) << 16);
      }
    }
    // e2 -> columns [128,192); SSL branch also contributes W3[64..127]
    {
      const float4* e4 = (const float4*)(e2p + h * 32);
      const float4* w4 = (const float4*)(W3 + 64 + h * 32);   // only used for SSL
      unsigned int* drow = (unsigned int*)(sA + (size_t)rl * A_STRIDE + 128 + h * 32);
      #pragma unroll
      for (int q = 0; q < 8; ++q) {
        float4 v = e4[q];
        if (branch == 0) {
          float4 w = w4[q];
          dot += v.x * w.x + v.y * w.y + v.z * w.z + v.w * w.w;
        }
        drow[q * 2 + 0] = (unsigned int)f2bf(v.x) | ((unsigned int)f2bf(v.y) << 16);
        drow[q * 2 + 1] = (unsigned int)f2bf(v.z) | ((unsigned int)f2bf(v.w) << 16);
      }
    }
    float tot = dot + __shfl_xor(dot, 1, 32);
    if (h == 0) {
      float x3 = loss * tot + b3;
      x3 = (x3 >= 0.f) ? x3 : a_pr * x3;                       // prelu
      x3buf[rl] = x3;
      xvals[(size_t)(by * 2 + 1) * S_TOTAL + grow] = x3;
    }
  }
  __syncthreads();

  // ---- GEMM phase: wave w -> rows [16w,16w+16), 6 N-tiles, 6 K-steps ----
  {
    const int wave = tid >> 5;
    const int lane = tid & 31;
    const int l16  = lane & 15;
    const int half = lane >> 4;
    const int m0   = wave * 16;

    v8f acc[6];
    #pragma unroll
    for (int n = 0; n < 6; ++n)
      #pragma unroll
      for (int v = 0; v < 8; ++v) acc[n][v] = 0.f;

    #pragma unroll
    for (int kt = 0; kt < 6; ++kt) {
      // A fragment: row = m0 + l16; K = kt*32 + half*8 + {0..7, 16..23}
      const int arow = m0 + l16;
      const int kbA  = kt * 32 + half * 8;
      v8bf alo = *(const v8bf*)(sA + (size_t)arow * A_STRIDE + kbA);
      v8bf ahi = *(const v8bf*)(sA + (size_t)arow * A_STRIDE + kbA + 16);
      v16bf a = __builtin_shufflevector(alo, ahi, 0,1,2,3,4,5,6,7,8,9,10,11,12,13,14,15);
      #pragma unroll
      for (int nt = 0; nt < 6; ++nt) {
        // B fragment: col = nt*16 + l16; K = kt*32 + half*8 + {0..7, 16..23}
        const int col = nt * 16 + l16;
        const int kbB = kt * 32 + half * 8;
        v8bf blo = *(const v8bf*)(sW + (size_t)col * W_STRIDE + kbB);
        v8bf bhi = *(const v8bf*)(sW + (size_t)col * W_STRIDE + kbB + 16);
        v16bf b = __builtin_shufflevector(blo, bhi, 0,1,2,3,4,5,6,7,8,9,10,11,12,13,14,15);
        acc[nt] = __builtin_amdgcn_wmma_f32_16x16x32_bf16(
            false, a, false, b, (short)0, acc[nt], false, false);
      }
    }

    // epilogue: bias + prelu + W2 dot; C layout: lane holds N=l16, M = v + 8*half
    float part[8];
    #pragma unroll
    for (int v = 0; v < 8; ++v) part[v] = 0.f;
    #pragma unroll
    for (int nt = 0; nt < 6; ++nt) {
      const float bb = b1[nt * 16 + l16];
      const float ww = W2[nt * 16 + l16];
      #pragma unroll
      for (int v = 0; v < 8; ++v) {
        float t = acc[nt][v] + bb;
        t = (t >= 0.f) ? t : a_pr * t;
        part[v] += t * ww;
      }
    }
    #pragma unroll
    for (int v = 0; v < 8; ++v) {
      part[v] += __shfl_xor(part[v], 1, 32);
      part[v] += __shfl_xor(part[v], 2, 32);
      part[v] += __shfl_xor(part[v], 4, 32);
      part[v] += __shfl_xor(part[v], 8, 32);   // reduce within 16-lane half
    }
    if (l16 == 0) {
      const float s3h = 13.856406460551018f;   // sqrt(3*H) = sqrt(192)
      #pragma unroll
      for (int v = 0; v < 8; ++v) {
        const int rl = m0 + half * 8 + v;
        float x1 = s3h * (part[v] + b2);
        x1buf[rl] = x1;
        xvals[(size_t)(by * 2) * S_TOTAL + (size_t)bx * ROWS + rl] = x1;
      }
    }
  }
  __syncthreads();

  // ---- per-block BN partials (deterministic tree) ----
  if (tid < 128) {
    float x1 = x1buf[tid], x3 = x3buf[tid];
    float s0 = x1, s1 = x1 * x1, s2 = x3, s3 = x3 * x3;
    #pragma unroll
    for (int m = 1; m < 32; m <<= 1) {
      s0 += __shfl_xor(s0, m, 32);
      s1 += __shfl_xor(s1, m, 32);
      s2 += __shfl_xor(s2, m, 32);
      s3 += __shfl_xor(s3, m, 32);
    }
    if ((tid & 31) == 0) {
      int w = tid >> 5;
      red4[w][0] = s0; red4[w][1] = s1; red4[w][2] = s2; red4[w][3] = s3;
    }
  }
  __syncthreads();
  if (tid == 0) {
    float4 t;
    t.x = red4[0][0] + red4[1][0] + red4[2][0] + red4[3][0];
    t.y = red4[0][1] + red4[1][1] + red4[2][1] + red4[3][1];
    t.z = red4[0][2] + red4[1][2] + red4[2][2] + red4[3][2];
    t.w = red4[0][3] + red4[1][3] + red4[2][3] + red4[3][3];
    partials[(size_t)by * NBLK + bx] = t;
  }
}

// ---------------------------------------------------------------------------
// Kernel 3: reduce 2048 partials per (branch,beh) -> mean / rstd for x1 & x3.
// ---------------------------------------------------------------------------
__global__ __launch_bounds__(256) void mwn_stats(const float4* __restrict__ partials,
                                                 float4* __restrict__ stats) {
  const int by = blockIdx.x;      // 0..5
  const int tid = threadIdx.x;
  __shared__ float buf[8][4];

  float s0 = 0.f, s1 = 0.f, s2 = 0.f, s3 = 0.f;
  for (int i = tid; i < NBLK; i += 256) {
    float4 p = partials[(size_t)by * NBLK + i];
    s0 += p.x; s1 += p.y; s2 += p.z; s3 += p.w;
  }
  #pragma unroll
  for (int m = 1; m < 32; m <<= 1) {
    s0 += __shfl_xor(s0, m, 32);
    s1 += __shfl_xor(s1, m, 32);
    s2 += __shfl_xor(s2, m, 32);
    s3 += __shfl_xor(s3, m, 32);
  }
  if ((tid & 31) == 0) {
    int w = tid >> 5;
    buf[w][0] = s0; buf[w][1] = s1; buf[w][2] = s2; buf[w][3] = s3;
  }
  __syncthreads();
  if (tid == 0) {
    float t0 = 0.f, t1 = 0.f, t2 = 0.f, t3 = 0.f;
    #pragma unroll
    for (int w = 0; w < 8; ++w) { t0 += buf[w][0]; t1 += buf[w][1]; t2 += buf[w][2]; t3 += buf[w][3]; }
    const float inv = 1.f / (float)S_TOTAL;
    float m1 = t0 * inv, v1 = t1 * inv - m1 * m1;
    float m3 = t2 * inv, v3 = t3 * inv - m3 * m3;
    float4 st;
    st.x = m1; st.y = rsqrtf(v1 + BN_EPS);
    st.z = m3; st.w = rsqrtf(v3 + BN_EPS);
    stats[by] = st;
  }
}

// ---------------------------------------------------------------------------
// Kernel 4: BN-normalize + sigmoid + combine into out[2][3][S].
// ---------------------------------------------------------------------------
__global__ __launch_bounds__(256) void mwn_final(const float* __restrict__ xvals,
                                                 const float4* __restrict__ stats,
                                                 float* __restrict__ out) {
  const int by = blockIdx.y;
  const int i  = blockIdx.x * 256 + threadIdx.x;
  const float4 st = stats[by];
  float x1 = xvals[(size_t)(by * 2 + 0) * S_TOTAL + i];
  float x3 = xvals[(size_t)(by * 2 + 1) * S_TOTAL + i];
  float y1 = 1.f / (1.f + __expf(-(x1 - st.x) * st.y));
  float y3 = 1.f / (1.f + __expf(-(x3 - st.z) * st.w));
  const int branch = by / 3, beh = by % 3;
  float r = (branch == 0) ? 0.5f * (y1 + y3) : (y1 + y3);
  out[(size_t)branch * 3 * S_TOTAL + (size_t)beh * S_TOTAL + i] = r;
}

// ---------------------------------------------------------------------------
extern "C" void kernel_launch(void* const* d_in, const int* in_sizes, int n_in,
                              void* d_out, int out_size, void* d_ws, size_t ws_size,
                              hipStream_t stream) {
  const float* infoNCE = (const float*)d_in[0];
  const float* blossm  = (const float*)d_in[1];
  const int*   stepi   = (const int*)d_in[2];
  const int*   uidxl   = (const int*)d_in[3];
  const float* uembs   = (const float*)d_in[4];
  const float* uemb    = (const float*)d_in[5];
  const float* W_ssl1  = (const float*)d_in[6];
  const float* b_ssl1  = (const float*)d_in[7];
  const float* W_ssl2  = (const float*)d_in[8];
  const float* b_ssl2  = (const float*)d_in[9];
  const float* W_ssl3  = (const float*)d_in[10];
  const float* b_ssl3  = (const float*)d_in[11];
  const float* W_rs1   = (const float*)d_in[12];
  const float* b_rs1   = (const float*)d_in[13];
  const float* W_rs2   = (const float*)d_in[14];
  const float* b_rs2   = (const float*)d_in[15];
  const float* W_rs3   = (const float*)d_in[16];
  const float* b_rs3   = (const float*)d_in[17];
  const float* prelu_a = (const float*)d_in[18];

  char* ws = (char*)d_ws;
  unsigned short* Wbf_ssl = (unsigned short*)(ws + 0);                  // 36864 B
  unsigned short* Wbf_rs  = (unsigned short*)(ws + 36864);              // 36864 B
  float*  xvals    = (float*)(ws + 73728);                              // 12*S*4 B
  float4* partials = (float4*)(ws + 73728 + (size_t)12 * S_TOTAL * 4);  // 6*2048*16 B
  float4* stats    = (float4*)((char*)partials + (size_t)6 * NBLK * sizeof(float4));

  wconv<<<dim3(72), dim3(256), 0, stream>>>(W_ssl1, W_rs1, Wbf_ssl, Wbf_rs);
  mwn_main<<<dim3(NBLK, 6), dim3(256), 0, stream>>>(
      infoNCE, blossm, stepi, uidxl, uembs, uemb, Wbf_ssl, Wbf_rs,
      b_ssl1, W_ssl2, b_ssl2, W_ssl3, b_ssl3,
      b_rs1, W_rs2, b_rs2, W_rs3, b_rs3, prelu_a, xvals, partials);
  mwn_stats<<<dim3(6), dim3(256), 0, stream>>>(partials, stats);
  mwn_final<<<dim3(S_TOTAL / 256, 6), dim3(256), 0, stream>>>(xvals, stats, (float*)d_out);
}
